// MLPPredictor_73400991089147
// MI455X (gfx1250) — matrix operand order repaired and checked by reading, not compiled, not run on hardware
//
#include <hip/hip_runtime.h>
#include <hip/hip_bf16.h>

// ---------------------------------------------------------------------------
// Types for CDNA5 WMMA (gfx1250, wave32)
// ---------------------------------------------------------------------------
typedef __attribute__((ext_vector_type(16))) __bf16 bf16x16;
typedef __attribute__((ext_vector_type(8)))  float  f32x8;

union ABFrag {
    uint4   q[2];   // two 16B loads
    bf16x16 v;      // 16 bf16 = 32B fragment
};

// ---------------------------------------------------------------------------
// Elementwise fp32 -> bf16 convert
// ---------------------------------------------------------------------------
__global__ void k_convert_bf16(const float* __restrict__ in,
                               __bf16* __restrict__ out, long long n) {
    long long i = (long long)blockIdx.x * blockDim.x + threadIdx.x;
    if (i < n) out[i] = (__bf16)in[i];
}

// fp32 W[K x N] (row-major) -> bf16 Wt[N x K] (row-major, i.e. transposed)
__global__ void k_transpose_bf16(const float* __restrict__ W,
                                 __bf16* __restrict__ Wt, int K, int N) {
    long long i = (long long)blockIdx.x * blockDim.x + threadIdx.x;
    long long total = (long long)K * N;
    if (i >= total) return;
    int n = (int)(i % N);
    int k = (int)(i / N);
    Wt[(long long)n * K + k] = (__bf16)W[i];
}

// ---------------------------------------------------------------------------
// Tiny vector MLP layer (cell networks): y[n] = act(sum_k x[k]*W[k,n] + b[n])
// ---------------------------------------------------------------------------
__global__ void k_vec_layer(const float* __restrict__ x,
                            const float* __restrict__ W,
                            const float* __restrict__ b,
                            float* __restrict__ y,
                            int K, int N, int relu) {
    int n = blockIdx.x * blockDim.x + threadIdx.x;
    if (n >= N) return;
    float s = b[n];
    for (int k = 0; k < K; ++k) s += x[k] * W[(long long)k * N + n];
    if (relu) s = fmaxf(s, 0.0f);
    y[n] = s;
}

// ---------------------------------------------------------------------------
// bf16 WMMA GEMM:  out[M x N] = act(A[M x K] @ Bt[N x K]^T + bias)
// A row-major bf16, Bt = transposed weight (row n holds column n, K contiguous).
//
// One wave computes a 64x64 block = 4x4 grid of 16x16 WMMA tiles:
//  - per 32-wide K-step: 4 A fragments + 4 B fragments feed 16 WMMAs
//    -> 32 FLOP/byte of fragment traffic (4x better than 1-tile waves)
//  - out-of-range M sub-tiles clamp their A row (loads stay valid, EXEC
//    stays all-ones for WMMA); only the stores are guarded.
//
// Fragment layouts per CDNA5 ISA (CDNA5 ISA):
//  A 16x32 : lane<16  -> M=lane,    a[0..7]=K0..7,  a[8..15]=K16..23
//            lane>=16 -> M=lane-16, a[0..7]=K8..15, a[8..15]=K24..31
//  B 32x16 : lane<16  -> N=lane,    b[0..7]=K0..7,  b[8..15]=K16..23
//            lane>=16 -> N=lane-16, b[0..7]=K8..15, b[8..15]=K24..31
//  C/D f32 : c[v]: lane<16 -> (M=v,   N=lane), lane>=16 -> (M=v+8, N=lane-16)
// ---------------------------------------------------------------------------
template<int RELU, int OUT_BF16>
__global__ __launch_bounds__(256)
void k_gemm_bf16(const __bf16* __restrict__ A,
                 const __bf16* __restrict__ Bt,
                 const float* __restrict__ bias,
                 void* __restrict__ outp,
                 int M, int N, int K) {
    const int wave = blockIdx.x * (blockDim.x >> 5) + (threadIdx.x >> 5);
    const int lane = threadIdx.x & 31;
    const int bn      = N >> 6;              // 64-wide N blocks (N % 64 == 0)
    const int bmCount = (M + 63) >> 6;       // 64-tall M blocks (clamped)
    if (wave >= bmCount * bn) return;        // wave-uniform: EXEC all-ones

    const int bm   = wave / bn;
    const int bcol = wave - bm * bn;
    const int m0 = bm << 6, n0 = bcol << 6;
    const int r  = lane & 15;
    const int hi = lane >> 4;                // half-wave select
    const int aOff = hi ? 8 : 0;             // A: interleaved K-halves
    const int bOff = hi ? 8 : 0;             // B: interleaved K-halves (match A)

    long long aRow[4], bRow[4];
#pragma unroll
    for (int i = 0; i < 4; ++i) {
        int row = m0 + i * 16 + r;
        if (row > M - 1) row = M - 1;        // clamp: valid loads, junk not stored
        aRow[i] = (long long)row * K;
    }
#pragma unroll
    for (int j = 0; j < 4; ++j)
        bRow[j] = (long long)(n0 + j * 16 + r) * K;

    f32x8 acc[4][4] = {};

    for (int k0 = 0; k0 < K; k0 += 32) {
        if (k0 + 128 < K) {                  // gfx1250 global_prefetch_b8 (WGP scope)
#pragma unroll
            for (int i = 0; i < 4; ++i) __builtin_prefetch(A  + aRow[i] + k0 + 128, 0, 3);
#pragma unroll
            for (int j = 0; j < 4; ++j) __builtin_prefetch(Bt + bRow[j] + k0 + 128, 0, 3);
        }
        ABFrag af[4], bf[4];
#pragma unroll
        for (int i = 0; i < 4; ++i) {
            af[i].q[0] = *(const uint4*)(A + aRow[i] + k0 + aOff);
            af[i].q[1] = *(const uint4*)(A + aRow[i] + k0 + 16 + aOff);
        }
#pragma unroll
        for (int j = 0; j < 4; ++j) {
            const __bf16* bp = Bt + bRow[j] + k0 + bOff;
            bf[j].q[0] = *(const uint4*)(bp);
            bf[j].q[1] = *(const uint4*)(bp + 16);
        }
#pragma unroll
        for (int i = 0; i < 4; ++i)
#pragma unroll
            for (int j = 0; j < 4; ++j)
                acc[i][j] = __builtin_amdgcn_wmma_f32_16x16x32_bf16(
                    false, af[i].v, false, bf[j].v, (short)0, acc[i][j], false, false);
    }

    float bv[4];
#pragma unroll
    for (int j = 0; j < 4; ++j) bv[j] = bias[n0 + j * 16 + r];

#pragma unroll
    for (int i = 0; i < 4; ++i) {
        const int rowBase = m0 + i * 16 + hi * 8;
#pragma unroll
        for (int v = 0; v < 8; ++v) {
            const int row = rowBase + v;
            if (row < M) {                   // store guard (post-WMMA divergence ok)
#pragma unroll
                for (int j = 0; j < 4; ++j) {
                    float x = acc[i][j][v] + bv[j];
                    if (RELU) x = fmaxf(x, 0.0f);
                    long long idx = (long long)row * N + (n0 + j * 16 + r);
                    if (OUT_BF16) ((__bf16*)outp)[idx] = (__bf16)x;
                    else          ((float*)outp)[idx]  = x;
                }
            }
        }
    }
}

// ---------------------------------------------------------------------------
// Edge feature gather: feat[e] = [n1[s],n2[s],n3[s],n1[d],n2[d],n3[d],c1,c2,c3]
// (all sources already relu'd bf16). 288 threads/block, one block per edge,
// each thread moves 8 bf16 (16B vector).
// ---------------------------------------------------------------------------
__global__ void k_gather_feat(const __bf16* __restrict__ n1,
                              const __bf16* __restrict__ n2,
                              const __bf16* __restrict__ n3,
                              const __bf16* __restrict__ cvec,  // 768 bf16
                              const int* __restrict__ src,
                              const int* __restrict__ dst,
                              __bf16* __restrict__ feat,
                              int e0, int ec) {
    int e = blockIdx.x;
    if (e >= ec) return;
    int t    = threadIdx.x;       // 0..287
    int elem = t << 3;            // *8 bf16
    int seg  = elem >> 8;         // 256-wide segment id 0..8
    int so   = elem & 255;
    long long sN = (long long)src[e0 + e] * 256;
    long long dN = (long long)dst[e0 + e] * 256;
    const __bf16* base;
    switch (seg) {
        case 0: base = n1 + sN; break;
        case 1: base = n2 + sN; break;
        case 2: base = n3 + sN; break;
        case 3: base = n1 + dN; break;
        case 4: base = n2 + dN; break;
        case 5: base = n3 + dN; break;
        default: base = cvec + (seg - 6) * 256; break;
    }
    *(uint4*)(feat + (long long)e * 2304 + elem) = *(const uint4*)(base + so);
}

// ---------------------------------------------------------------------------
// Final layer: out[e] = dot(h3[e, 0:256], w4) + b4   (one wave per edge)
// ---------------------------------------------------------------------------
__global__ void k_final_dot(const float* __restrict__ h3,
                            const float* __restrict__ w4,
                            const float* __restrict__ b4,
                            float* __restrict__ out,
                            int e0, int ec) {
    int wave = blockIdx.x * (blockDim.x >> 5) + (threadIdx.x >> 5);
    int lane = threadIdx.x & 31;
    if (wave >= ec) return;
    const float* row = h3 + (long long)wave * 256;
    float s = 0.0f;
#pragma unroll
    for (int i = 0; i < 8; ++i) s += row[lane + i * 32] * w4[lane + i * 32];
    for (int off = 16; off; off >>= 1) s += __shfl_xor(s, off, 32);
    if (lane == 0) out[e0 + wave] = s + b4[0];
}

// ---------------------------------------------------------------------------
// Host orchestration
// ---------------------------------------------------------------------------
static inline int cdiv(long long a, long long b) { return (int)((a + b - 1) / b); }

static void launch_gemm(hipStream_t s, const __bf16* A, const __bf16* Bt,
                        const float* bias, void* out, int M, int N, int K,
                        bool bf16out) {
    int waves = ((M + 63) / 64) * (N / 64);
    int grid  = cdiv(waves, 8);
    if (bf16out) k_gemm_bf16<1, 1><<<grid, 256, 0, s>>>(A, Bt, bias, out, M, N, K);
    else         k_gemm_bf16<1, 0><<<grid, 256, 0, s>>>(A, Bt, bias, out, M, N, K);
}

static void cvt(hipStream_t s, const float* in, __bf16* out, long long n) {
    k_convert_bf16<<<cdiv(n, 256), 256, 0, s>>>(in, out, n);
}

static void tpose(hipStream_t s, const float* W, __bf16* Wt, int K, int N) {
    k_transpose_bf16<<<cdiv((long long)K * N, 256), 256, 0, s>>>(W, Wt, K, N);
}

extern "C" void kernel_launch(void* const* d_in, const int* in_sizes, int n_in,
                              void* d_out, int out_size, void* d_ws, size_t ws_size,
                              hipStream_t stream) {
    (void)in_sizes; (void)n_in; (void)out_size; (void)ws_size;

    enum { NN = 10000, D = 256, H = 512, E = 90000,
           FEAT = 2304, F1 = 2048, F2 = 1024, F3 = 256, CHUNK = 4608 };

    // ---- inputs (setup_inputs dict order) ----
    const float* h       = (const float*)d_in[0];
    const float* spec_h  = (const float*)d_in[1];
    const float* orig_h  = (const float*)d_in[2];
    const float* cellf   = (const float*)d_in[3];
    const float* orig_cf = (const float*)d_in[4];
    const float* net_cf  = (const float*)d_in[5];
    const int*   src     = (const int*)d_in[6];
    const int*   dst     = (const int*)d_in[7];
    const float *d1w1 = (const float*)d_in[8],  *d1b1 = (const float*)d_in[9];
    const float *d1w2 = (const float*)d_in[10], *d1b2 = (const float*)d_in[11];
    const float *d2w1 = (const float*)d_in[12], *d2b1 = (const float*)d_in[13];
    const float *d2w2 = (const float*)d_in[14], *d2b2 = (const float*)d_in[15];
    const float *d3w1 = (const float*)d_in[16], *d3b1 = (const float*)d_in[17];
    const float *d3w2 = (const float*)d_in[18], *d3b2 = (const float*)d_in[19];
    const float *c1w1 = (const float*)d_in[20], *c1b1 = (const float*)d_in[21];
    const float *c1w2 = (const float*)d_in[22], *c1b2 = (const float*)d_in[23];
    const float *c2w1 = (const float*)d_in[24], *c2b1 = (const float*)d_in[25];
    const float *c2w2 = (const float*)d_in[26], *c2b2 = (const float*)d_in[27];
    const float *c3w1 = (const float*)d_in[28], *c3b1 = (const float*)d_in[29];
    const float *c3w2 = (const float*)d_in[30], *c3b2 = (const float*)d_in[31];
    const float *fw1  = (const float*)d_in[32], *fb1  = (const float*)d_in[33];
    const float *fw2  = (const float*)d_in[34], *fb2  = (const float*)d_in[35];
    const float *fw3  = (const float*)d_in[36], *fb3  = (const float*)d_in[37];
    const float *fw4  = (const float*)d_in[38], *fb4  = (const float*)d_in[39];
    float* out = (float*)d_out;

    // ---- workspace carve (256B aligned) ----
    char* base = (char*)d_ws;
    size_t off = 0;
    auto carve = [&](size_t bytes) -> void* {
        void* p = base + off;
        off += (bytes + 255) & ~(size_t)255;
        return p;
    };
    __bf16* xin   = (__bf16*)carve((size_t)NN * D * 2);      // node input bf16 (reused)
    __bf16* nhid  = (__bf16*)carve((size_t)NN * H * 2);      // node hidden bf16 (reused)
    __bf16* n1b   = (__bf16*)carve((size_t)NN * D * 2);      // relu'd node outputs
    __bf16* n2b   = (__bf16*)carve((size_t)NN * D * 2);
    __bf16* n3b   = (__bf16*)carve((size_t)NN * D * 2);
    __bf16* d1w1t = (__bf16*)carve((size_t)H * D * 2);       // bf16 transposed weights
    __bf16* d1w2t = (__bf16*)carve((size_t)D * H * 2);
    __bf16* d2w1t = (__bf16*)carve((size_t)H * D * 2);
    __bf16* d2w2t = (__bf16*)carve((size_t)D * H * 2);
    __bf16* d3w1t = (__bf16*)carve((size_t)H * D * 2);
    __bf16* d3w2t = (__bf16*)carve((size_t)D * H * 2);
    __bf16* fw1t  = (__bf16*)carve((size_t)F1 * FEAT * 2);
    __bf16* fw2t  = (__bf16*)carve((size_t)F2 * F1 * 2);
    __bf16* fw3t  = (__bf16*)carve((size_t)F3 * F2 * 2);
    float*  chid  = (float*)carve(2048 * 4);                 // cell hidden fp32
    float*  cf32  = (float*)carve(768 * 4);                  // relu'd c1|c2|c3 fp32
    __bf16* cbf   = (__bf16*)carve(768 * 2);                 // bf16 copy for gather
    __bf16* feat  = (__bf16*)carve((size_t)CHUNK * FEAT * 2);
    __bf16* h1    = (__bf16*)carve((size_t)CHUNK * F1 * 2);
    __bf16* h2    = (__bf16*)carve((size_t)CHUNK * F2 * 2);
    float*  h3    = (float*)carve((size_t)CHUNK * F3 * 4);

    // ---- 1) weights -> bf16 transposed ----
    tpose(stream, d1w1, d1w1t, D, H);   tpose(stream, d1w2, d1w2t, H, D);
    tpose(stream, d2w1, d2w1t, D, H);   tpose(stream, d2w2, d2w2t, H, D);
    tpose(stream, d3w1, d3w1t, D, H);   tpose(stream, d3w2, d3w2t, H, D);
    tpose(stream, fw1,  fw1t, FEAT, F1);
    tpose(stream, fw2,  fw2t, F1,  F2);
    tpose(stream, fw3,  fw3t, F2,  F3);

    // ---- 2) cell networks (tiny fp32; relu folded into final output) ----
    k_vec_layer<<<cdiv(2048, 128), 128, 0, stream>>>(cellf,   c1w1, c1b1, chid,       1024, 2048, 1);
    k_vec_layer<<<cdiv(256,  128), 128, 0, stream>>>(chid,    c1w2, c1b2, cf32 + 0,   2048, 256,  1);
    k_vec_layer<<<cdiv(512,  128), 128, 0, stream>>>(orig_cf, c2w1, c2b1, chid,       512,  512,  1);
    k_vec_layer<<<cdiv(256,  128), 128, 0, stream>>>(chid,    c2w2, c2b2, cf32 + 256, 512,  256,  1);
    k_vec_layer<<<cdiv(512,  128), 128, 0, stream>>>(net_cf,  c3w1, c3b1, chid,       256,  512,  1);
    k_vec_layer<<<cdiv(256,  128), 128, 0, stream>>>(chid,    c3w2, c3b2, cf32 + 512, 512,  256,  1);
    cvt(stream, cf32, cbf, 768);

    // ---- 3) node MLPs (WMMA bf16; relu folded into final output) ----
    const float* nin[3]   = { orig_h, spec_h, h };
    const __bf16* w1t[3]  = { d1w1t, d2w1t, d3w1t };
    const __bf16* w2t[3]  = { d1w2t, d2w2t, d3w2t };
    const float* b1p[3]   = { d1b1, d2b1, d3b1 };
    const float* b2p[3]   = { d1b2, d2b2, d3b2 };
    __bf16* nout[3]       = { n1b, n2b, n3b };
    for (int i = 0; i < 3; ++i) {
        cvt(stream, nin[i], xin, (long long)NN * D);
        launch_gemm(stream, xin,  w1t[i], b1p[i], nhid,    NN, H, D, true);
        launch_gemm(stream, nhid, w2t[i], b2p[i], nout[i], NN, D, H, true);
    }

    // ---- 4) FC network over edges, chunked ----
    for (int e0 = 0; e0 < E; e0 += CHUNK) {
        int ec = (E - e0 < (int)CHUNK) ? (E - e0) : (int)CHUNK;  // always %16==0
        k_gather_feat<<<ec, 288, 0, stream>>>(n1b, n2b, n3b, cbf, src, dst,
                                              feat, e0, ec);
        launch_gemm(stream, feat, fw1t, fb1, h1, ec, F1, FEAT, true);
        launch_gemm(stream, h1,   fw2t, fb2, h2, ec, F2, F1,   true);
        launch_gemm(stream, h2,   fw3t, fb3, h3, ec, F3, F2,   false); // fp32 out, relu
        k_final_dot<<<cdiv(ec, 8), 256, 0, stream>>>(h3, fw4, fb4, out, e0, ec);
    }
}
